// cross_att_5995774345291
// MI455X (gfx1250) — matrix operand order repaired and checked by reading, not compiled, not run on hardware
//
#include <hip/hip_runtime.h>
#include <hip/hip_bf16.h>
#include <math.h>
#include <stdint.h>

// ---------------------------------------------------------------------------
// Problem constants (from the reference)
// ---------------------------------------------------------------------------
#define B_   2
#define D1_  32
#define D2_  1024
#define C_   256
#define NH_  8
#define LN_EPS 1e-5f
#define ROWS_TOTAL (B_ * D1_ * D2_)   // 65536 total sequence rows
#define SLICES     (B_ * D1_)         // 64

typedef __attribute__((ext_vector_type(16))) __bf16 v16bf;
typedef __attribute__((ext_vector_type(2)))  __bf16 v2bf;
typedef __attribute__((ext_vector_type(8)))  float  v8f;

// A-matrix (16x32 bf16) per-lane K index pattern (contiguous, matches B):
//   VGPR r: elements (2r,2r+1)  -> K = 16*half + 2r   (+0,+1)
__device__ __forceinline__ int aColBase(int r, int hh) {
    return hh * 16 + 2 * r;
}

// CDNA5 async direct-to-LDS copy (VGLOBAL GV mode, tracked by ASYNCcnt).
// lds_addr: 32-bit LDS byte address (generic pointer low word, per ISA 10.2),
// gaddr   : 64-bit global byte address.
__device__ __forceinline__ void async_copy_b128(unsigned lds_addr,
                                                unsigned long long gaddr) {
    asm volatile("global_load_async_to_lds_b128 %0, %1, off"
                 :
                 : "v"(lds_addr), "v"(gaddr)
                 : "memory");
}

__device__ __forceinline__ void wait_asynccnt0() {
    asm volatile("s_wait_asynccnt 0x0" ::: "memory");
}

// Wave-local LDS RAW fence (LDS ops are DScnt-tracked, in order per wave).
__device__ __forceinline__ void wait_dscnt0() {
    asm volatile("s_wait_dscnt 0x0" ::: "memory");
}

// Stage one 32-key step: K tile (32x256, contiguous 16 KB slab) and V^T tile
// (256 chans x 32 keys, 64B run per chan) into the given LDS double-buffer
// half, using fire-and-forget async copies (16 x b128 per thread).
__device__ __forceinline__ void stage_tiles(const __bf16* Kslice,
                                            const __bf16* Vtslice,
                                            __bf16* sKb, __bf16* sVb,
                                            int kt0, unsigned tid) {
    const unsigned skb = (unsigned)(uintptr_t)(void*)sKb;
    const unsigned svb = (unsigned)(uintptr_t)(void*)sVb;
    const unsigned long long ksrc =
        (unsigned long long)(uintptr_t)(Kslice + (size_t)kt0 * C_);
    const unsigned long long vsrc =
        (unsigned long long)(uintptr_t)(Vtslice + kt0);
#pragma unroll
    for (int i = 0; i < 8; ++i) {
        const unsigned j = tid + 128 * i;           // 16B chunk index
        async_copy_b128(skb + j * 16, ksrc + (unsigned long long)j * 16);
    }
#pragma unroll
    for (int i = 0; i < 8; ++i) {
        const unsigned j  = tid + 128 * i;          // 16B chunk index
        const unsigned ch = j >> 2;
        const unsigned sq = (j & 3) * 8;            // bf16 elements
        async_copy_b128(svb + (ch * 32 + sq) * 2,
                        vsrc + ((unsigned long long)ch * D2_ + sq) * 2);
    }
}

// ---------------------------------------------------------------------------
// Kernel 1: fused Q/K/V projection (X * W^T + b), fp32 in -> bf16 out.
// One wave computes a 16-row x 256-col block of all three outputs.
// Q is pre-scaled by 1/sqrt(D2/NUM_HEAD); V is stored transposed
// ([slice][chan][seq]) so kernel 2's B-fragments are contiguous.
// ---------------------------------------------------------------------------
__global__ __launch_bounds__(128)
void qkv_proj_wmma(const float* __restrict__ x, const float* __restrict__ x2,
                   const float* __restrict__ Wq, const float* __restrict__ bq,
                   const float* __restrict__ Wk, const float* __restrict__ bk,
                   const float* __restrict__ Wv, const float* __restrict__ bv,
                   __bf16* __restrict__ Qws, __bf16* __restrict__ Kws,
                   __bf16* __restrict__ Vtws)
{
    const int lane = threadIdx.x & 31;
    const int wv   = threadIdx.x >> 5;
    const int hh   = lane >> 4;     // half of the wave (B/C layout split)
    const int lm   = lane & 15;
    const int rowBase = (blockIdx.x * 4 + wv) * 16;   // grid = 1024 blocks
    const int slice = rowBase / D2_;
    const int srow  = rowBase % D2_;

    const float qscale = 1.0f / sqrtf((float)D2_ / (float)NH_);

    // A fragments for the 16 rows (row = rowBase + lm), all 256 channels.
    v16bf xa[8], x2a[8];
    const float* xrow  = x  + (size_t)(rowBase + lm) * C_;
    const float* x2row = x2 + (size_t)(rowBase + lm) * C_;
#pragma unroll
    for (int c = 0; c < 8; ++c) {
#pragma unroll
        for (int r = 0; r < 8; ++r) {
            const int k = c * 32 + aColBase(r, hh);
            float2 fx  = *(const float2*)(xrow  + k);
            float2 fx2 = *(const float2*)(x2row + k);
            xa [c][2*r] = (__bf16)fx.x;  xa [c][2*r+1] = (__bf16)fx.y;
            x2a[c][2*r] = (__bf16)fx2.x; x2a[c][2*r+1] = (__bf16)fx2.y;
        }
    }

#pragma unroll 1
    for (int nt = 0; nt < 16; ++nt) {
        v8f qacc = {}, kacc = {}, vacc = {};
        const int n = nt * 16 + lm;     // output channel owned by this lane
#pragma unroll
        for (int c = 0; c < 8; ++c) {
            // B fragment of W^T: B[k][n] = W[n][k]; 16 contiguous fp32/lane.
            const float* wqp = Wq + (size_t)n * C_ + c * 32 + hh * 16;
            const float* wkp = Wk + (size_t)n * C_ + c * 32 + hh * 16;
            const float* wvp = Wv + (size_t)n * C_ + c * 32 + hh * 16;
            v16bf qb, kb, vb;
#pragma unroll
            for (int e = 0; e < 16; e += 4) {
                float4 a = *(const float4*)(wqp + e);
                qb[e]=(__bf16)a.x; qb[e+1]=(__bf16)a.y; qb[e+2]=(__bf16)a.z; qb[e+3]=(__bf16)a.w;
                float4 b = *(const float4*)(wkp + e);
                kb[e]=(__bf16)b.x; kb[e+1]=(__bf16)b.y; kb[e+2]=(__bf16)b.z; kb[e+3]=(__bf16)b.w;
                float4 d = *(const float4*)(wvp + e);
                vb[e]=(__bf16)d.x; vb[e+1]=(__bf16)d.y; vb[e+2]=(__bf16)d.z; vb[e+3]=(__bf16)d.w;
            }
            qacc = __builtin_amdgcn_wmma_f32_16x16x32_bf16(false, x2a[c], false, qb, (short)0, qacc, false, false);
            kacc = __builtin_amdgcn_wmma_f32_16x16x32_bf16(false, xa[c],  false, kb, (short)0, kacc, false, false);
            vacc = __builtin_amdgcn_wmma_f32_16x16x32_bf16(false, xa[c],  false, vb, (short)0, vacc, false, false);
        }
        const float bqv = bq[n], bkv = bk[n], bvv = bv[n];
#pragma unroll
        for (int r = 0; r < 8; ++r) {
            const int m = hh * 8 + r;   // C/D layout: row = 8*half + vgpr
            Qws[(size_t)(rowBase + m) * C_ + n] = (__bf16)((qacc[r] + bqv) * qscale);
            Kws[(size_t)(rowBase + m) * C_ + n] = (__bf16)(kacc[r] + bkv);
            Vtws[((size_t)slice * C_ + n) * D2_ + (srow + m)] = (__bf16)(vacc[r] + bvv);
        }
    }
}

// ---------------------------------------------------------------------------
// Kernel 2: flash-style attention + out-proj collapse + residual + LayerNorm.
// Workgroup = 4 waves, 64 query rows; 32-key steps. K/V^T tiles are staged in
// a double-buffered LDS region via CDNA5 async direct-to-LDS copies, with the
// next step's copies issued before this step's compute (copy/compute overlap,
// ASYNCcnt-ordered). Online softmax per row (rows live in 16-lane halves of
// the wave in the WMMA C layout -> shfl_xor reductions with masks 1..8).
// ---------------------------------------------------------------------------
__global__ __launch_bounds__(128)
void attn_ln_wmma(const float* __restrict__ x,
                  const __bf16* __restrict__ Qws, const __bf16* __restrict__ Kws,
                  const __bf16* __restrict__ Vtws,
                  const float* __restrict__ w_out, const float* __restrict__ b_out,
                  const float* __restrict__ gamma, const float* __restrict__ beta,
                  float* __restrict__ out)
{
    __shared__ __align__(32) __bf16 sK [2][32 * 256];   // [buf][krow][chan] 32 KB
    __shared__ __align__(32) __bf16 sVt[2][256 * 32];   // [buf][chan][krow] 32 KB
    __shared__ __align__(32) __bf16 sP [4 * 16 * 32];   // per-wave P tile    4 KB

    const int lane = threadIdx.x & 31;
    const int wv   = threadIdx.x >> 5;
    const int hh   = lane >> 4;
    const int lm   = lane & 15;
    const int slice = blockIdx.x >> 4;           // 64 slices
    const int qblk  = blockIdx.x & 15;           // 16 blocks of 64 q rows
    const int qBase = qblk * 64 + wv * 16;       // within slice
    const size_t rowGlob = (size_t)slice * D2_ + qBase;

    const __bf16* Kslice  = Kws  + (size_t)slice * D2_ * C_;
    const __bf16* Vtslice = Vtws + (size_t)slice * C_  * D2_;

    // Q A-fragments (bf16, already scaled by 1/kd).
    v16bf qa[8];
    const __bf16* qrow = Qws + (rowGlob + lm) * C_;
#pragma unroll
    for (int c = 0; c < 8; ++c) {
#pragma unroll
        for (int r = 0; r < 8; ++r) {
            const int k = c * 32 + aColBase(r, hh);
            v2bf p = *(const v2bf*)(qrow + k);
            qa[c][2*r] = p[0]; qa[c][2*r+1] = p[1];
        }
    }

    v8f o[16];
#pragma unroll
    for (int nt = 0; nt < 16; ++nt) { v8f z = {}; o[nt] = z; }
    float mrow[8], lrow[8];
#pragma unroll
    for (int r = 0; r < 8; ++r) { mrow[r] = -3.0e38f; lrow[r] = 0.0f; }

    __bf16* pw = sP + wv * (16 * 32);

    // Prologue: stage step 0 into buffer 0.
    stage_tiles(Kslice, Vtslice, sK[0], sVt[0], 0, threadIdx.x);

#pragma unroll 1
    for (int step = 0; step < 32; ++step) {
        const int cur = step & 1;
        const __bf16* sKc = sK[cur];
        const __bf16* sVc = sVt[cur];

        // Current buffer's async copies (issued last iteration) have landed.
        wait_asynccnt0();
        __syncthreads();

        // Kick off the next step's staging; it overlaps with this step's
        // compute. The buffer it writes was last read in step-1, and all
        // waves passed that step's trailing barrier already.
        if (step + 1 < 32)
            stage_tiles(Kslice, Vtslice, sK[cur ^ 1], sVt[cur ^ 1],
                        (step + 1) * 32, threadIdx.x);

        // --- S = Q * K^T (two 16x16 tiles covering 32 keys) ---
        v8f st[2];
#pragma unroll
        for (int sub = 0; sub < 2; ++sub) {
            v8f acc = {};
#pragma unroll
            for (int c = 0; c < 8; ++c) {
                // B frag: n = key col (lm), K = channel (16*half + e)
                v16bf kb = *(const v16bf*)(sKc + (sub * 16 + lm) * 256 + c * 32 + hh * 16);
                acc = __builtin_amdgcn_wmma_f32_16x16x32_bf16(false, qa[c], false, kb, (short)0, acc, false, false);
            }
            st[sub] = acc;
        }

        // --- online softmax over these 32 columns ---
        float corr[8];
#pragma unroll
        for (int r = 0; r < 8; ++r) {
            float tmax = fmaxf(st[0][r], st[1][r]);
            tmax = fmaxf(tmax, __shfl_xor(tmax, 1, 32));
            tmax = fmaxf(tmax, __shfl_xor(tmax, 2, 32));
            tmax = fmaxf(tmax, __shfl_xor(tmax, 4, 32));
            tmax = fmaxf(tmax, __shfl_xor(tmax, 8, 32));
            const float mnew = fmaxf(mrow[r], tmax);
            const float p0 = __expf(st[0][r] - mnew);
            const float p1 = __expf(st[1][r] - mnew);
            float rs = p0 + p1;
            rs += __shfl_xor(rs, 1, 32);
            rs += __shfl_xor(rs, 2, 32);
            rs += __shfl_xor(rs, 4, 32);
            rs += __shfl_xor(rs, 8, 32);
            const float cf = __expf(mrow[r] - mnew);
            lrow[r] = lrow[r] * cf + rs;
            mrow[r] = mnew;
            corr[r] = cf;
            st[0][r] = p0; st[1][r] = p1;
        }
#pragma unroll
        for (int nt = 0; nt < 16; ++nt)
#pragma unroll
            for (int r = 0; r < 8; ++r) o[nt][r] *= corr[r];

        // --- re-layout P: C layout -> row-major LDS -> A fragment ---
        // sP is private to this wave: a wave-local DScnt fence is enough
        // (LDS ops are in order per wave), no workgroup barrier needed.
#pragma unroll
        for (int r = 0; r < 8; ++r) {
            const int m = hh * 8 + r;
            pw[m * 32 + lm]      = (__bf16)st[0][r];
            pw[m * 32 + 16 + lm] = (__bf16)st[1][r];
        }
        wait_dscnt0();

        v16bf pa;
#pragma unroll
        for (int r = 0; r < 8; ++r) {
            const int col = aColBase(r, hh);
            v2bf pp = *(const v2bf*)(pw + lm * 32 + col);
            pa[2*r] = pp[0]; pa[2*r+1] = pp[1];
        }

        // --- O += P * V (B frag from transposed V: contiguous per lane) ---
#pragma unroll
        for (int nt = 0; nt < 16; ++nt) {
            v16bf vb = *(const v16bf*)(sVc + (nt * 16 + lm) * 32 + hh * 16);
            o[nt] = __builtin_amdgcn_wmma_f32_16x16x32_bf16(false, pa, false, vb, (short)0, o[nt], false, false);
        }

        // All waves must finish reading this buffer before it is re-staged
        // in the next iteration.
        __syncthreads();
    }

    // --- epilogue: out-proj collapse, residual, LayerNorm over C ---
    float sumw = 0.0f;
#pragma unroll
    for (int i = 0; i < NH_; ++i) sumw += w_out[i];
    const float bo = b_out[0];

#pragma unroll 1
    for (int r = 0; r < 8; ++r) {
        const int m = hh * 8 + r;
        const size_t row = rowGlob + m;
        const float invl = 1.0f / lrow[r];
        float h[16];
        float s1 = 0.0f;
#pragma unroll
        for (int nt = 0; nt < 16; ++nt) {
            const int cch = nt * 16 + lm;
            const float hv = x[row * C_ + cch] + o[nt][r] * invl * sumw + bo;
            h[nt] = hv;
            s1 += hv;
        }
        s1 += __shfl_xor(s1, 1, 32);
        s1 += __shfl_xor(s1, 2, 32);
        s1 += __shfl_xor(s1, 4, 32);
        s1 += __shfl_xor(s1, 8, 32);
        const float mu = s1 * (1.0f / C_);
        float s2 = 0.0f;
#pragma unroll
        for (int nt = 0; nt < 16; ++nt) {
            const float d = h[nt] - mu;
            s2 += d * d;
        }
        s2 += __shfl_xor(s2, 1, 32);
        s2 += __shfl_xor(s2, 2, 32);
        s2 += __shfl_xor(s2, 4, 32);
        s2 += __shfl_xor(s2, 8, 32);
        const float rstd = rsqrtf(s2 * (1.0f / C_) + LN_EPS);
#pragma unroll
        for (int nt = 0; nt < 16; ++nt) {
            const int cch = nt * 16 + lm;
            out[row * C_ + cch] = (h[nt] - mu) * rstd * gamma[cch] + beta[cch];
        }
    }
}

// ---------------------------------------------------------------------------
// Launch: workspace holds Q (bf16 row-major), K (bf16 row-major),
// V^T (bf16 [slice][chan][seq]) — 3 * 65536 * 256 * 2B = ~100.7 MB.
// ---------------------------------------------------------------------------
extern "C" void kernel_launch(void* const* d_in, const int* in_sizes, int n_in,
                              void* d_out, int out_size, void* d_ws, size_t ws_size,
                              hipStream_t stream) {
    (void)in_sizes; (void)n_in; (void)out_size; (void)ws_size;

    const float* x     = (const float*)d_in[0];
    const float* x2    = (const float*)d_in[1];
    const float* Wq    = (const float*)d_in[2];
    const float* bq    = (const float*)d_in[3];
    const float* Wk    = (const float*)d_in[4];
    const float* bk    = (const float*)d_in[5];
    const float* Wv    = (const float*)d_in[6];
    const float* bv    = (const float*)d_in[7];
    const float* w_out = (const float*)d_in[8];
    const float* b_out = (const float*)d_in[9];
    const float* gamma = (const float*)d_in[10];
    const float* beta  = (const float*)d_in[11];
    float* out = (float*)d_out;

    __bf16* Qws = (__bf16*)d_ws;
    __bf16* Kws = Qws + (size_t)ROWS_TOTAL * C_;
    __bf16* Vt  = Kws + (size_t)ROWS_TOTAL * C_;

    // 65536 rows / (4 waves * 16 rows) = 1024 blocks of 128 threads.
    qkv_proj_wmma<<<dim3(1024), dim3(128), 0, stream>>>(
        x, x2, Wq, bq, Wk, bk, Wv, bv, Qws, Kws, Vt);

    // 64 slices * 16 q-blocks = 1024 blocks of 128 threads.
    attn_ln_wmma<<<dim3(1024), dim3(128), 0, stream>>>(
        x, Qws, Kws, Vt, w_out, b_out, gamma, beta, out);
}
